// OptionA_48455821033919
// MI455X (gfx1250) — compile-verified, matrix operand order checked by reading, not run on hardware
//
#include <hip/hip_runtime.h>
#include <hip/hip_bf16.h>
#include <stdint.h>
#include <stddef.h>

// ---------------- problem constants (match reference) ----------------
constexpr int NN       = 131072;          // nodes
constexpr int NE       = 524288;          // edges
constexpr int E2       = NE + NN;         // edges + self loops
constexpr int NB       = 1024;            // graphs
constexpr int NODE_DIM = 35;
constexpr int EDGE_DIM = 13;
constexpr int HEADS    = 4;
constexpr int HD       = 32;              // head dim
constexpr int HID      = 128;             // HEADS*HD
constexpr int COND_IN  = 11;              // 3 + BOX_EMB
constexpr int MLP_H    = 256;
constexpr float BN_EPS = 1e-5f;
constexpr float NEG_SLOPE = 0.2f;

typedef __attribute__((ext_vector_type(16))) __bf16 v16bf;
typedef __attribute__((ext_vector_type(8)))  __bf16 v8bf;
typedef __attribute__((ext_vector_type(8)))  float  v8f;

static inline int cdiv(int a, int b) { return (a + b - 1) / b; }

// ---------------- utility kernels ----------------
__global__ void k_zero_f(float* p, size_t n) {
  size_t i = (size_t)blockIdx.x * blockDim.x + threadIdx.x;
  if (i < n) p[i] = 0.0f;
}
__global__ void k_fill_u32(unsigned* p, unsigned v, size_t n) {
  size_t i = (size_t)blockIdx.x * blockDim.x + threadIdx.x;
  if (i < n) p[i] = v;
}
__global__ void k_cast_bf16(const float* __restrict__ in, __bf16* __restrict__ out, size_t n) {
  size_t i = (size_t)blockIdx.x * blockDim.x + threadIdx.x;
  if (i < n) out[i] = (__bf16)in[i];
}
// Bt[n*K+k] = (bf16) W[k*N+n]
__global__ void k_transpose_cast(const float* __restrict__ W, __bf16* __restrict__ Bt, int K, int N) {
  int i = blockIdx.x * blockDim.x + threadIdx.x;
  if (i >= K * N) return;
  int k = i / N, n = i % N;
  Bt[(size_t)n * K + k] = (__bf16)W[(size_t)k * N + n];
}

// ---- fragment load helper: 16-bit A/B ISA layout for one 32-wide K step ----
// lane = half*16 + l16 ; row pointer must point at (tile row l16) start of K range.
__device__ __forceinline__ v16bf load_frag(const __bf16* row, int k0, int half) {
  v8bf lo = *(const v8bf*)(row + k0 + half * 8);
  v8bf hi = *(const v8bf*)(row + k0 + 16 + half * 8);
  return __builtin_shufflevector(lo, hi, 0,1,2,3,4,5,6,7,8,9,10,11,12,13,14,15);
}

// ---------------- generic WMMA GEMM: C[M,N] = A[M,K] @ B[K,N] (+bias)(+relu) ----------------
// A: bf16 row-major [M,K]; Bt: bf16 row-major [N,K]. K%32==0, M,N%16==0.
__global__ void k_gemm_bf16_wmma(const __bf16* __restrict__ A,
                                 const __bf16* __restrict__ Bt,
                                 const float* __restrict__ bias,
                                 float* __restrict__ C,
                                 int M, int N, int K, int relu) {
  int lane = threadIdx.x & 31;
  int wave = threadIdx.x >> 5;
  int tileN = blockIdx.x * 4 + wave;
  int tileM = blockIdx.y;
  if (tileN * 16 >= N || tileM * 16 >= M) return;
  int half = lane >> 4;
  int l16  = lane & 15;
  const __bf16* arow = A  + (size_t)(tileM * 16 + l16) * K;
  const __bf16* brow = Bt + (size_t)(tileN * 16 + l16) * K;

  v8f acc = {0.f, 0.f, 0.f, 0.f, 0.f, 0.f, 0.f, 0.f};
  for (int k0 = 0; k0 < K; k0 += 32) {
    v16bf a = load_frag(arow, k0, half);
    v16bf b = load_frag(brow, k0, half);
    acc = __builtin_amdgcn_wmma_f32_16x16x32_bf16(false, a, false, b, (short)0, acc, false, false);
  }
  int cn = tileN * 16 + l16;
  int cm = tileM * 16 + half * 8;
  float bv = bias ? bias[cn] : 0.0f;
#pragma unroll
  for (int v = 0; v < 8; ++v) {
    float val = acc[v] + bv;
    if (relu) val = fmaxf(val, 0.0f);
    C[(size_t)(cm + v) * N + cn] = val;
  }
}

// ---------------- fused dual GEMM for GAT layers: xl = A@Wl, xr = A@Wr ----------------
// M=NN, N=K=HID=128 fixed. B fragments (both matrices, all 4 K-steps) are hoisted into
// registers once per wave; wave then grid-strides over M tiles -> ~1 b128 load per WMMA.
__global__ void k_gat_lr_gemm(const __bf16* __restrict__ A,
                              const __bf16* __restrict__ BtL,
                              const __bf16* __restrict__ BtR,
                              float* __restrict__ xl, float* __restrict__ xr) {
  int lane = threadIdx.x & 31;
  int wave = threadIdx.x >> 5;
  int tileN = blockIdx.x * 4 + wave;          // 0..7
  int half = lane >> 4;
  int l16  = lane & 15;
  const __bf16* blrow = BtL + (size_t)(tileN * 16 + l16) * HID;
  const __bf16* brrow = BtR + (size_t)(tileN * 16 + l16) * HID;
  v16bf bl[4], br[4];
#pragma unroll
  for (int ks = 0; ks < 4; ++ks) {
    bl[ks] = load_frag(blrow, ks * 32, half);
    br[ks] = load_frag(brrow, ks * 32, half);
  }
  int cn = tileN * 16 + l16;
  for (int tm = blockIdx.y; tm < NN / 16; tm += gridDim.y) {
    const __bf16* arow = A + (size_t)(tm * 16 + l16) * HID;
    v8f accl = {0.f, 0.f, 0.f, 0.f, 0.f, 0.f, 0.f, 0.f};
    v8f accr = {0.f, 0.f, 0.f, 0.f, 0.f, 0.f, 0.f, 0.f};
#pragma unroll
    for (int ks = 0; ks < 4; ++ks) {
      v16bf a = load_frag(arow, ks * 32, half);
      accl = __builtin_amdgcn_wmma_f32_16x16x32_bf16(false, a, false, bl[ks], (short)0, accl, false, false);
      accr = __builtin_amdgcn_wmma_f32_16x16x32_bf16(false, a, false, br[ks], (short)0, accr, false, false);
    }
    int cm = tm * 16 + half * 8;
#pragma unroll
    for (int v = 0; v < 8; ++v) {
      xl[(size_t)(cm + v) * HID + cn] = accl[v];
      xr[(size_t)(cm + v) * HID + cn] = accr[v];
    }
  }
}

// ---------------- input projection: h = x @ Wi + bi   (K=35, VALU) ----------------
__global__ void k_input_proj(const float* __restrict__ x, const float* __restrict__ W,
                             const float* __restrict__ b, float* __restrict__ h) {
  size_t i = (size_t)blockIdx.x * blockDim.x + threadIdx.x;
  if (i >= (size_t)NN * HID) return;
  int n = (int)(i / HID), c = (int)(i % HID);
  const float* xr = x + (size_t)n * NODE_DIM;
  float acc = b[c];
#pragma unroll
  for (int j = 0; j < NODE_DIM; ++j) acc += xr[j] * W[j * HID + c];
  h[i] = acc;
}

// ---------------- self-loop attr: deg + sum of in-edge attrs ----------------
__global__ void k_loop_accum(const int* __restrict__ ei, const float* __restrict__ ea,
                             float* __restrict__ loop_sum, float* __restrict__ deg) {
  int e = blockIdx.x * blockDim.x + threadIdx.x;
  if (e >= NE) return;
  int dst = ei[NE + e];
  atomicAdd(&deg[dst], 1.0f);
  const float* a = ea + (size_t)e * EDGE_DIM;
#pragma unroll
  for (int j = 0; j < EDGE_DIM; ++j) atomicAdd(&loop_sum[dst * EDGE_DIM + j], a[j]);
}
__global__ void k_build_ea2_idx(const int* __restrict__ ei, const float* __restrict__ ea,
                                const float* __restrict__ loop_sum, const float* __restrict__ deg,
                                float* __restrict__ ea2, int* __restrict__ src2, int* __restrict__ dst2) {
  int e = blockIdx.x * blockDim.x + threadIdx.x;
  if (e >= E2) return;
  if (e < NE) {
    src2[e] = ei[e]; dst2[e] = ei[NE + e];
    const float* a = ea + (size_t)e * EDGE_DIM;
#pragma unroll
    for (int j = 0; j < EDGE_DIM; ++j) ea2[(size_t)e * EDGE_DIM + j] = a[j];
  } else {
    int n = e - NE;
    src2[e] = n; dst2[e] = n;
    float d = fmaxf(deg[n], 1.0f);
#pragma unroll
    for (int j = 0; j < EDGE_DIM; ++j)
      ea2[(size_t)e * EDGE_DIM + j] = loop_sum[n * EDGE_DIM + j] / d;
  }
}

// ---------------- ordered-key float atomic max ----------------
__device__ __forceinline__ unsigned fkey(float f) {
  unsigned u = __float_as_uint(f);
  return (u & 0x80000000u) ? ~u : (u | 0x80000000u);
}
__device__ __forceinline__ float fdec(unsigned k) {
  return (k & 0x80000000u) ? __uint_as_float(k & 0x7FFFFFFFu) : __uint_as_float(~k);
}

// ---------------- GATv2 score: one thread per (edge, head) ----------------
__global__ void k_gat_score(const float* __restrict__ xl, const float* __restrict__ xr,
                            const float* __restrict__ ea2, const float* __restrict__ We,
                            const float* __restrict__ att,
                            const int* __restrict__ src2, const int* __restrict__ dst2,
                            float* __restrict__ score, unsigned* __restrict__ smax) {
  int t = blockIdx.x * blockDim.x + threadIdx.x;
  if (t >= E2 * HEADS) return;
  int e = t >> 2, h = t & 3;
  int s = src2[e], d = dst2[e];
  float ea[EDGE_DIM];
#pragma unroll
  for (int j = 0; j < EDGE_DIM; ++j) ea[j] = ea2[(size_t)e * EDGE_DIM + j];
  const float* xls = xl + (size_t)s * HID + h * HD;
  const float* xrd = xr + (size_t)d * HID + h * HD;
  float acc = 0.0f;
#pragma unroll 4
  for (int k = 0; k < HD; ++k) {
    float ed = 0.0f;
#pragma unroll
    for (int j = 0; j < EDGE_DIM; ++j) ed += ea[j] * We[j * HID + h * HD + k];
    float m = xls[k] + xrd[k] + ed;
    float sv = (m > 0.0f) ? m : NEG_SLOPE * m;
    acc += sv * att[h * HD + k];
  }
  score[t] = acc;
  atomicMax(&smax[d * HEADS + h], fkey(acc));
}

// ---------------- exp + denominator ----------------
__global__ void k_gat_exp(const int* __restrict__ dst2, const unsigned* __restrict__ smax,
                          float* __restrict__ score, float* __restrict__ den) {
  int t = blockIdx.x * blockDim.x + threadIdx.x;
  if (t >= E2 * HEADS) return;
  int e = t >> 2, h = t & 3;
  int d = dst2[e];
  float ex = __expf(score[t] - fdec(smax[d * HEADS + h]));
  score[t] = ex;
  atomicAdd(&den[d * HEADS + h], ex);
}

// ---------------- scatter alpha * xl[src] into out : one thread per (edge, channel) ----------------
__global__ void k_gat_scatter(const float* __restrict__ xl, const int* __restrict__ src2,
                              const int* __restrict__ dst2, const float* __restrict__ score,
                              const float* __restrict__ den, float* __restrict__ out, int concat) {
  size_t t = (size_t)blockIdx.x * blockDim.x + threadIdx.x;
  if (t >= (size_t)E2 * HID) return;
  int e = (int)(t >> 7);          // / HID
  int c = (int)(t & (HID - 1));   // channel 0..127 (coalesced within wave)
  int h = c >> 5;
  int s = src2[e], d = dst2[e];
  float alpha = score[(size_t)e * HEADS + h] / den[d * HEADS + h];
  float v = alpha * xl[(size_t)s * HID + c];
  if (concat) atomicAdd(&out[(size_t)d * HID + c], v);
  else        atomicAdd(&out[(size_t)d * HD + (c & (HD - 1))], 0.25f * v);  // mean over heads
}

// ---------------- BatchNorm stats: blockDim.x == C ----------------
__global__ void k_bn_stats(const float* __restrict__ out, float* __restrict__ stats, int C) {
  int c = threadIdx.x;
  float s = 0.0f, sq = 0.0f;
  for (int r = blockIdx.x; r < NN; r += gridDim.x) {
    float v = out[(size_t)r * C + c];
    s += v; sq += v * v;
  }
  atomicAdd(&stats[c], s);
  atomicAdd(&stats[C + c], sq);
}

// ---------------- BN + ELU + identity residual (concat layers) ----------------
__global__ void k_bn_elu_res(const float* __restrict__ outacc, const float* __restrict__ stats,
                             const float* __restrict__ g, const float* __restrict__ bb,
                             const float* __restrict__ hold, float* __restrict__ hnew) {
  size_t i = (size_t)blockIdx.x * blockDim.x + threadIdx.x;
  if (i >= (size_t)NN * HID) return;
  int c = (int)(i % HID);
  float mu  = stats[c] / (float)NN;
  float var = stats[HID + c] / (float)NN - mu * mu;
  float nv  = (outacc[i] - mu) * rsqrtf(var + BN_EPS) * g[c] + bb[c];
  float el  = (nv > 0.0f) ? nv : expm1f(nv);
  hnew[i] = el + hold[i];
}

// ---------------- BN + ELU + linear-skip residual (last layer, C=32) ----------------
__global__ void k_bn_elu_skip(const float* __restrict__ outacc, const float* __restrict__ stats,
                              const float* __restrict__ g, const float* __restrict__ bb,
                              const float* __restrict__ hold, const float* __restrict__ Wskip,
                              float* __restrict__ hnew) {
  size_t i = (size_t)blockIdx.x * blockDim.x + threadIdx.x;
  if (i >= (size_t)NN * HD) return;
  int n = (int)(i / HD), c = (int)(i % HD);
  float mu  = stats[c] / (float)NN;
  float var = stats[HD + c] / (float)NN - mu * mu;
  float nv  = (outacc[i] - mu) * rsqrtf(var + BN_EPS) * g[c] + bb[c];
  float el  = (nv > 0.0f) ? nv : expm1f(nv);
  const float* hr = hold + (size_t)n * HID;
  float sk = 0.0f;
#pragma unroll 8
  for (int k = 0; k < HID; ++k) sk += hr[k] * Wskip[k * HD + c];
  hnew[i] = el + sk;
}

// ---------------- pooling ----------------
__global__ void k_pool_cnt(const int* __restrict__ batch, float* __restrict__ cnt) {
  int n = blockIdx.x * blockDim.x + threadIdx.x;
  if (n < NN) atomicAdd(&cnt[batch[n]], 1.0f);
}
__global__ void k_pool_sum(const float* __restrict__ h2, const int* __restrict__ batch,
                           float* __restrict__ gsum) {
  size_t i = (size_t)blockIdx.x * blockDim.x + threadIdx.x;
  if (i >= (size_t)NN * HID) return;
  int n = (int)(i / HID), c = (int)(i % HID);
  atomicAdd(&gsum[(size_t)batch[n] * HID + c], h2[i]);
}
__global__ void k_pool_div(const float* __restrict__ gsum, const float* __restrict__ cnt,
                           float* __restrict__ hh) {   // hh is [NB, 160]
  int i = blockIdx.x * blockDim.x + threadIdx.x;
  if (i >= NB * HID) return;
  int b = i / HID, c = i % HID;
  hh[(size_t)b * 160 + c] = gsum[i] / fmaxf(cnt[b], 1.0f);
}

// ---------------- conditioning MLP ----------------
__global__ void k_cond1(const float* __restrict__ ph, const float* __restrict__ temp,
                        const float* __restrict__ rec, const int* __restrict__ box_idx,
                        const float* __restrict__ box_emb, const float* __restrict__ W,
                        const float* __restrict__ bia, float* __restrict__ hc1) {
  int i = blockIdx.x * blockDim.x + threadIdx.x;
  if (i >= NB * 64) return;
  int b = i / 64, o = i % 64;
  float c[COND_IN];
  c[0] = ph[b]; c[1] = temp[b]; c[2] = rec[b];
  const float* be = box_emb + box_idx[b] * 8;
#pragma unroll
  for (int j = 0; j < 8; ++j) c[3 + j] = be[j];
  float acc = bia[o];
#pragma unroll
  for (int j = 0; j < COND_IN; ++j) acc += c[j] * W[j * 64 + o];
  hc1[i] = fmaxf(acc, 0.0f);
}
__global__ void k_cond2(const float* __restrict__ hc1, const float* __restrict__ W,
                        const float* __restrict__ bia, float* __restrict__ hh) {
  int i = blockIdx.x * blockDim.x + threadIdx.x;
  if (i >= NB * 32) return;
  int b = i / 32, o = i % 32;
  const float* r = hc1 + (size_t)b * 64;
  float acc = bia[o];
#pragma unroll 16
  for (int j = 0; j < 64; ++j) acc += r[j] * W[j * 32 + o];
  hh[(size_t)b * 160 + HID + o] = acc;
}

// ---------------- final linear [128 -> 1] ----------------
__global__ void k_mlp3(const float* __restrict__ m2, const float* __restrict__ w,
                       const float* __restrict__ b, float* __restrict__ out) {
  int bi = blockIdx.x * blockDim.x + threadIdx.x;
  if (bi >= NB) return;
  const float* r = m2 + (size_t)bi * 128;
  float acc = b[0];
#pragma unroll 16
  for (int k = 0; k < 128; ++k) acc += r[k] * w[k];
  out[bi] = acc;
}

// ---------------- host side ----------------
static char* ws_alloc(char*& cur, size_t bytes) {
  uintptr_t u = (uintptr_t)cur;
  u = (u + 255) & ~(uintptr_t)255;
  char* r = (char*)u;
  cur = r + bytes;
  return r;
}

extern "C" void kernel_launch(void* const* d_in, const int* in_sizes, int n_in,
                              void* d_out, int out_size, void* d_ws, size_t ws_size,
                              hipStream_t stream) {
  (void)in_sizes; (void)n_in; (void)out_size; (void)ws_size;
  // ---- inputs (setup_inputs dict order; params pytree: dicts sorted by key) ----
  const float* x        = (const float*)d_in[0];
  const int*   ei       = (const int*)  d_in[1];
  const float* ea       = (const float*)d_in[2];
  const int*   batch    = (const int*)  d_in[3];
  const float* ph       = (const float*)d_in[4];
  const float* temp     = (const float*)d_in[5];
  const int*   box_idx  = (const int*)  d_in[6];
  const float* rec      = (const float*)d_in[7];
  const float* box_emb  = (const float*)d_in[8];
  const float* cond1_b  = (const float*)d_in[9];
  const float* cond1_w  = (const float*)d_in[10];
  const float* cond2_b  = (const float*)d_in[11];
  const float* cond2_w  = (const float*)d_in[12];
  const float* inp_b    = (const float*)d_in[13];
  const float* inp_w    = (const float*)d_in[14];
  // layers 0..2: We,Wl,Wr,att,bias,bn_b,bn_g at 15+7*l; layer 3 adds skip
  const float* mlp1_b   = (const float*)d_in[44];
  const float* mlp1_w   = (const float*)d_in[45];
  const float* mlp2_b   = (const float*)d_in[46];
  const float* mlp2_w   = (const float*)d_in[47];
  const float* mlp3_b   = (const float*)d_in[48];
  const float* mlp3_w   = (const float*)d_in[49];
  const float* pg_b     = (const float*)d_in[50];
  const float* pg_w     = (const float*)d_in[51];
  float* outp = (float*)d_out;

  // ---- workspace carve ----
  char* cur = (char*)d_ws;
  float*    h      = (float*)   ws_alloc(cur, (size_t)NN * HID * 4);
  float*    hn     = (float*)   ws_alloc(cur, (size_t)NN * HID * 4);
  __bf16*   hbf    = (__bf16*)  ws_alloc(cur, (size_t)NN * HID * 2);
  float*    xl     = (float*)   ws_alloc(cur, (size_t)NN * HID * 4);
  float*    xr     = (float*)   ws_alloc(cur, (size_t)NN * HID * 4);
  float*    outacc = (float*)   ws_alloc(cur, (size_t)NN * HID * 4);
  float*    ea2    = (float*)   ws_alloc(cur, (size_t)E2 * EDGE_DIM * 4);
  float*    lsum   = (float*)   ws_alloc(cur, (size_t)NN * EDGE_DIM * 4);
  float*    deg    = (float*)   ws_alloc(cur, (size_t)NN * 4);
  int*      src2   = (int*)     ws_alloc(cur, (size_t)E2 * 4);
  int*      dst2   = (int*)     ws_alloc(cur, (size_t)E2 * 4);
  float*    score  = (float*)   ws_alloc(cur, (size_t)E2 * HEADS * 4);
  unsigned* smax   = (unsigned*)ws_alloc(cur, (size_t)NN * HEADS * 4);
  float*    den    = (float*)   ws_alloc(cur, (size_t)NN * HEADS * 4);
  float*    stats  = (float*)   ws_alloc(cur, 2 * HID * 4);
  __bf16*   wTa    = (__bf16*)  ws_alloc(cur, (size_t)HID * HID * 2);
  __bf16*   wTb    = (__bf16*)  ws_alloc(cur, (size_t)HID * HID * 2);
  __bf16*   wTc    = (__bf16*)  ws_alloc(cur, (size_t)65536 * 2);
  float*    gsum   = (float*)   ws_alloc(cur, (size_t)NB * HID * 4);
  float*    cnt    = (float*)   ws_alloc(cur, (size_t)NB * 4);
  float*    hc1    = (float*)   ws_alloc(cur, (size_t)NB * 64 * 4);
  float*    hh     = (float*)   ws_alloc(cur, (size_t)NB * 160 * 4);
  __bf16*   hhbf   = (__bf16*)  ws_alloc(cur, (size_t)NB * 160 * 2);
  float*    m1     = (float*)   ws_alloc(cur, (size_t)NB * MLP_H * 4);
  __bf16*   m1bf   = (__bf16*)  ws_alloc(cur, (size_t)NB * MLP_H * 2);
  float*    m2     = (float*)   ws_alloc(cur, (size_t)NB * 128 * 4);
  __bf16*   m2bf   = (__bf16*)  ws_alloc(cur, (size_t)NB * 128 * 2);
  float*    h2     = (float*)   ws_alloc(cur, (size_t)NN * HID * 4);

  const int T = 256;

  // ---- self-loop edge_attr + combined edge list ----
  hipLaunchKernelGGL(k_zero_f, dim3(cdiv(NN, T)), dim3(T), 0, stream, deg, (size_t)NN);
  hipLaunchKernelGGL(k_zero_f, dim3(cdiv(NN * EDGE_DIM, T)), dim3(T), 0, stream, lsum, (size_t)NN * EDGE_DIM);
  hipLaunchKernelGGL(k_loop_accum, dim3(cdiv(NE, T)), dim3(T), 0, stream, ei, ea, lsum, deg);
  hipLaunchKernelGGL(k_build_ea2_idx, dim3(cdiv(E2, T)), dim3(T), 0, stream, ei, ea, lsum, deg, ea2, src2, dst2);

  // ---- input projection ----
  hipLaunchKernelGGL(k_input_proj, dim3(cdiv(NN * HID, T)), dim3(T), 0, stream, x, inp_w, inp_b, h);

  // ---- GATv2 layers ----
  for (int l = 0; l < 4; ++l) {
    int base = 15 + 7 * l;
    const float* We   = (const float*)d_in[base + 0];
    const float* Wl   = (const float*)d_in[base + 1];
    const float* Wr   = (const float*)d_in[base + 2];
    const float* att  = (const float*)d_in[base + 3];
    const float* bn_b = (const float*)d_in[base + 5];
    const float* bn_g = (const float*)d_in[base + 6];
    const float* skip = (l == 3) ? (const float*)d_in[base + 7] : nullptr;
    int concat = (l < 3);
    int C = concat ? HID : HD;

    // h -> bf16, weights -> bf16 transposed
    hipLaunchKernelGGL(k_cast_bf16, dim3(cdiv(NN * HID, T)), dim3(T), 0, stream, h, hbf, (size_t)NN * HID);
    hipLaunchKernelGGL(k_transpose_cast, dim3(cdiv(HID * HID, T)), dim3(T), 0, stream, Wl, wTa, HID, HID);
    hipLaunchKernelGGL(k_transpose_cast, dim3(cdiv(HID * HID, T)), dim3(T), 0, stream, Wr, wTb, HID, HID);
    // fused xl = h@Wl, xr = h@Wr (WMMA, B fragments hoisted, M grid-strided)
    hipLaunchKernelGGL(k_gat_lr_gemm, dim3(2, 1024), dim3(128), 0, stream, hbf, wTa, wTb, xl, xr);

    // attention softmax over in-edges
    hipLaunchKernelGGL(k_fill_u32, dim3(cdiv(NN * HEADS, T)), dim3(T), 0, stream, smax, 0u, (size_t)NN * HEADS);
    hipLaunchKernelGGL(k_zero_f,  dim3(cdiv(NN * HEADS, T)), dim3(T), 0, stream, den, (size_t)NN * HEADS);
    hipLaunchKernelGGL(k_gat_score, dim3(cdiv(E2 * HEADS, T)), dim3(T), 0, stream,
                       xl, xr, ea2, We, att, src2, dst2, score, smax);
    hipLaunchKernelGGL(k_gat_exp, dim3(cdiv(E2 * HEADS, T)), dim3(T), 0, stream, dst2, smax, score, den);
    hipLaunchKernelGGL(k_zero_f, dim3(cdiv(NN * C, T)), dim3(T), 0, stream, outacc, (size_t)NN * C);
    hipLaunchKernelGGL(k_gat_scatter, dim3(cdiv(E2 * HID, T)), dim3(T), 0, stream,
                       xl, src2, dst2, score, den, outacc, concat);

    // BatchNorm (GAT bias is a constant shift -> cancelled by BN mean subtraction) + ELU + residual
    hipLaunchKernelGGL(k_zero_f, dim3(1), dim3(T), 0, stream, stats, (size_t)(2 * C));
    hipLaunchKernelGGL(k_bn_stats, dim3(512), dim3(C), 0, stream, outacc, stats, C);
    if (concat) {
      hipLaunchKernelGGL(k_bn_elu_res, dim3(cdiv(NN * HID, T)), dim3(T), 0, stream,
                         outacc, stats, bn_g, bn_b, h, hn);
    } else {
      hipLaunchKernelGGL(k_bn_elu_skip, dim3(cdiv(NN * HD, T)), dim3(T), 0, stream,
                         outacc, stats, bn_g, bn_b, h, skip, hn);
    }
    float* t = h; h = hn; hn = t;   // h now holds layer output
  }

  // ---- post-GNN linear [N,32] @ [32,128] + b (WMMA, K=32) ----
  hipLaunchKernelGGL(k_cast_bf16, dim3(cdiv(NN * HD, T)), dim3(T), 0, stream, h, hbf, (size_t)NN * HD);
  hipLaunchKernelGGL(k_transpose_cast, dim3(cdiv(HD * HID, T)), dim3(T), 0, stream, pg_w, wTc, HD, HID);
  hipLaunchKernelGGL(k_gemm_bf16_wmma, dim3(cdiv(HID, 64), NN / 16), dim3(128), 0, stream,
                     hbf, wTc, pg_b, h2, NN, HID, HD, 0);

  // ---- global mean pool ----
  hipLaunchKernelGGL(k_zero_f, dim3(cdiv(NB * HID, T)), dim3(T), 0, stream, gsum, (size_t)NB * HID);
  hipLaunchKernelGGL(k_zero_f, dim3(cdiv(NB, T)), dim3(T), 0, stream, cnt, (size_t)NB);
  hipLaunchKernelGGL(k_pool_cnt, dim3(cdiv(NN, T)), dim3(T), 0, stream, batch, cnt);
  hipLaunchKernelGGL(k_pool_sum, dim3(cdiv(NN * HID, T)), dim3(T), 0, stream, h2, batch, gsum);
  hipLaunchKernelGGL(k_pool_div, dim3(cdiv(NB * HID, T)), dim3(T), 0, stream, gsum, cnt, hh);

  // ---- conditioning ----
  hipLaunchKernelGGL(k_cond1, dim3(cdiv(NB * 64, T)), dim3(T), 0, stream,
                     ph, temp, rec, box_idx, box_emb, cond1_w, cond1_b, hc1);
  hipLaunchKernelGGL(k_cond2, dim3(cdiv(NB * 32, T)), dim3(T), 0, stream, hc1, cond2_w, cond2_b, hh);

  // ---- MLP head: [B,160]@[160,256] relu -> [B,256]@[256,128] relu -> [B,128]@[128,1] ----
  hipLaunchKernelGGL(k_cast_bf16, dim3(cdiv(NB * 160, T)), dim3(T), 0, stream, hh, hhbf, (size_t)NB * 160);
  hipLaunchKernelGGL(k_transpose_cast, dim3(cdiv(160 * MLP_H, T)), dim3(T), 0, stream, mlp1_w, wTc, 160, MLP_H);
  hipLaunchKernelGGL(k_gemm_bf16_wmma, dim3(cdiv(MLP_H, 64), NB / 16), dim3(128), 0, stream,
                     hhbf, wTc, mlp1_b, m1, NB, MLP_H, 160, 1);
  hipLaunchKernelGGL(k_cast_bf16, dim3(cdiv(NB * MLP_H, T)), dim3(T), 0, stream, m1, m1bf, (size_t)NB * MLP_H);
  hipLaunchKernelGGL(k_transpose_cast, dim3(cdiv(MLP_H * 128, T)), dim3(T), 0, stream, mlp2_w, wTc, MLP_H, 128);
  hipLaunchKernelGGL(k_gemm_bf16_wmma, dim3(cdiv(128, 64), NB / 16), dim3(128), 0, stream,
                     m1bf, wTc, mlp2_b, m2, NB, 128, MLP_H, 1);
  hipLaunchKernelGGL(k_mlp3, dim3(cdiv(NB, T)), dim3(T), 0, stream, m2, mlp3_w, mlp3_b, outp);
}
